// LSTM_20633022890193
// MI455X (gfx1250) — compile-verified, matrix operand order checked by reading, not run on hardware
//
#include <hip/hip_runtime.h>
#include <hip/hip_bf16.h>
#include <stdint.h>

// Problem dims
#define I_DIM 256
#define H_DIM 1024
#define O_DIM 256
#define B_DIM 64
#define T_DIM 512
#define G4H   (4 * H_DIM)

typedef __attribute__((ext_vector_type(16))) __bf16 v16bf;
typedef __attribute__((ext_vector_type(8)))  float  v8f;
typedef __attribute__((ext_vector_type(4)))  int    v4i;

// Address-space qualified v4i for the async-to-LDS builtin:
// param0 is a global (AS1) v4i*, param1 the LDS (AS3) v4i*.
typedef __attribute__((address_space(1))) v4i AS1v4i;
typedef __attribute__((address_space(3))) v4i AS3v4i;

#if __has_builtin(__builtin_amdgcn_global_load_async_to_lds_b128)
#define USE_ASYNC_B 1
#else
#define USE_ASYNC_B 0
#endif

static __device__ __forceinline__ float fast_sigmoid(float x) {
    return 1.0f / (1.0f + __expf(-x));
}

// Wait until at most N async(-to-LDS) ops remain outstanding (per wave).
template <int N>
static __device__ __forceinline__ void wait_asynccnt() {
#if USE_ASYNC_B
#if __has_builtin(__builtin_amdgcn_s_wait_asynccnt)
    __builtin_amdgcn_s_wait_asynccnt(N);
#else
    asm volatile("s_wait_asynccnt %0" :: "i"(N) : "memory");
#endif
#endif
}

// Move one 16B segment global -> LDS (async on gfx1250, sync fallback).
static __device__ __forceinline__ void copy16_to_lds(const __hip_bfloat16* gp,
                                                     __hip_bfloat16* lp) {
#if USE_ASYNC_B
    __builtin_amdgcn_global_load_async_to_lds_b128(
        (AS1v4i*)gp, (AS3v4i*)lp, 0, 0);
#else
    *(uint4*)lp = *(const uint4*)gp;
#endif
}

// ---------------------------------------------------------------------------
// f32 -> bf16 convert
// ---------------------------------------------------------------------------
__global__ void cvt_f32_bf16(__hip_bfloat16* __restrict__ dst,
                             const float* __restrict__ src, int n) {
    int i = blockIdx.x * blockDim.x + threadIdx.x;
    if (i < n) dst[i] = __float2bfloat16(src[i]);
}

// Pack W_ih (4H x ipart) and W_hh (4H x H) into one bf16 matrix [4H][kc],
// kc = ipart + H, so the recurrent GEMM uses A = [x_t ; h_{t-1}] directly.
__global__ void pack_wcat_bf16(__hip_bfloat16* __restrict__ dst,
                               const float* __restrict__ w_ih,
                               const float* __restrict__ w_hh,
                               int ipart, int kc, int n) {
    int i = blockIdx.x * blockDim.x + threadIdx.x;
    if (i >= n) return;
    int row = i / kc;
    int col = i - row * kc;
    float v = (col < ipart) ? w_ih[(size_t)row * ipart + col]
                            : w_hh[(size_t)row * H_DIM + (col - ipart)];
    dst[i] = __float2bfloat16(v);
}

// ---------------------------------------------------------------------------
// Fused LSTM step:  gates = [x_t ; h_{t-1}] @ Wcat^T  (+ biases), then cell.
//
// Grid: H/16 = 64 WGs; block 128 threads (4 wave32).
//   WG owns hidden cols j0..j0+15; wave w owns batch rows 16w..16w+15 and
//   accumulates all four gate tiles (i,f,g,o) in registers, so the LSTM
//   cell update fuses in-register after the K loop.
//
// B staging: all 4 waves need identical B tiles, so each 4 KB K-chunk of
// Wcat (4 gates x 16 rows x 32 K, bf16) is fetched ONCE per WG with
// GLOBAL_LOAD_ASYNC_TO_LDS_B128 into a 4-slot LDS ring (3 chunks in flight,
// 2 async instructions per wave per chunk -> uniform s_wait_asynccnt 4),
// gated by a workgroup barrier; waves read fragments via ds_load_b128.
// Tail issues are clamped to the last chunk but target ring slots that are
// never read again, keeping the wait threshold branch-free.
//
// Register layouts per ISA 7.12.2 (bf16): element e of v16bf at lane l maps
// to K = (e/8)*16 + (l/16)*8 + e%8 -> two 16B chunks [ka..ka+7],[kb..kb+7].
// C layout: vgpr r -> row r + 8*(lane>=16), lane%16 -> col.
// ---------------------------------------------------------------------------
__global__ __launch_bounds__(128)
void lstm_step_wmma(const __hip_bfloat16* __restrict__ xpart, int x_stride, int ipart,
                    const __hip_bfloat16* __restrict__ hprev,
                    const __hip_bfloat16* __restrict__ W, int kc,
                    const float* __restrict__ b_ih,
                    const float* __restrict__ b_hh,
                    float* __restrict__ cstate,
                    __hip_bfloat16* __restrict__ hout) {
    __shared__ __align__(16) __hip_bfloat16 Bsh[4][4 * 16 * 32];  // 4 x 4 KB ring

    const int tid  = threadIdx.x;
    const int lane = tid & 31;
    const int wave = tid >> 5;
    const int m0   = wave * 16;
    const int j0   = blockIdx.x * 16;
    const int nn   = lane & 15;
    const int hi   = lane >> 4;          // 0: lanes 0-15, 1: lanes 16-31
    const int mA   = m0 + nn;            // A-matrix row this lane supplies
    const int nchunk = kc >> 5;

    // A-source row bases (x part, and h part shifted so `+ k0` works directly)
    const __hip_bfloat16* xrow = xpart + (size_t)mA * x_stride;
    const __hip_bfloat16* hrow = hprev + (size_t)mA * H_DIM - ipart;

    // This thread's two fixed 16B segments of a B chunk (straight-line, no
    // lane-divergent loop): segment s covers W row (gate*16+r), 16B piece q.
    const int s0 = tid;                      // 0..127
    const int s1 = tid + 128;                // 128..255
    const int rr0 = s0 >> 2, q0 = s0 & 3;    // rr = gate*16 + r
    const int rr1 = s1 >> 2, q1 = s1 & 3;
    const __hip_bfloat16* wrow0 =
        W + (size_t)((rr0 >> 4) * H_DIM + j0 + (rr0 & 15)) * kc + q0 * 8;
    const __hip_bfloat16* wrow1 =
        W + (size_t)((rr1 >> 4) * H_DIM + j0 + (rr1 & 15)) * kc + q1 * 8;
    __hip_bfloat16* lds0 = &Bsh[0][rr0 * 32 + q0 * 8];
    __hip_bfloat16* lds1 = &Bsh[0][rr1 * 32 + q1 * 8];

    // Issue chunk index cc into ring slot `slot`.
    auto issue_chunk = [&](int cc, int slot) {
        const int k0 = cc << 5;
        const int lo = slot * (4 * 16 * 32);
        copy16_to_lds(wrow0 + k0, lds0 + lo);
        copy16_to_lds(wrow1 + k0, lds1 + lo);
    };

    v8f acc[4] = {};                     // i, f, g, o accumulators (f32)

    issue_chunk(0, 0);                   // nchunk >= 40, no clamping needed
    issue_chunk(1, 1);
    issue_chunk(2, 2);

    for (int c = 0; c < nchunk; ++c) {
        const int k0 = c << 5;

        // A fragment: issue global loads before the async wait/barrier so
        // their latency hides behind the B pipeline. Uniform x/h select
        // (ipart is a multiple of 32, chunks never straddle).
        const __hip_bfloat16* ar = (k0 < ipart) ? xrow : hrow;
        union { uint4 u[2]; v16bf v; } A;
        A.u[0] = *(const uint4*)(ar + k0 + hi * 8);
        A.u[1] = *(const uint4*)(ar + k0 + 16 + hi * 8);

        // 2 chunks (4 async ops) stay in flight -> chunk c has landed.
        wait_asynccnt<4>();
        __syncthreads();                 // chunk c visible to all waves

        const __hip_bfloat16* bbase = Bsh[c & 3];
#pragma unroll
        for (int g = 0; g < 4; ++g) {
            const __hip_bfloat16* bp = bbase + (g * 16 + nn) * 32;
            union { uint4 u[2]; v16bf v; } Bv;
            Bv.u[0] = *(const uint4*)(bp + hi * 8);
            Bv.u[1] = *(const uint4*)(bp + 16 + hi * 8);
            acc[g] = __builtin_amdgcn_wmma_f32_16x16x32_bf16(
                false, A.v, false, Bv.v, (short)0, acc[g], false, false);
        }
        __syncthreads();                 // all waves done reading slot (c-1)&3's era

        // Keep issue rate constant: clamp chunk index in the tail; the
        // clamped issues land in slots never read again (distance-4 ring).
        int cc = c + 3;
        if (cc > nchunk - 1) cc = nchunk - 1;
        issue_chunk(cc, (c + 3) & 3);
    }

    // Fused LSTM cell update (per-lane elements are matched i/f/g/o)
    const int j  = j0 + nn;
    const float bi = b_ih[0 * H_DIM + j] + b_hh[0 * H_DIM + j];
    const float bf = b_ih[1 * H_DIM + j] + b_hh[1 * H_DIM + j];
    const float bg = b_ih[2 * H_DIM + j] + b_hh[2 * H_DIM + j];
    const float bo = b_ih[3 * H_DIM + j] + b_hh[3 * H_DIM + j];

#pragma unroll
    for (int r = 0; r < 8; ++r) {
        const int m = m0 + r + hi * 8;
        const float iv = fast_sigmoid(acc[0][r] + bi);
        const float fv = fast_sigmoid(acc[1][r] + bf);
        const float gv = tanhf(acc[2][r] + bg);
        const float ov = fast_sigmoid(acc[3][r] + bo);
        const size_t idx = (size_t)m * H_DIM + j;
        const float cn = fv * cstate[idx] + iv * gv;
        cstate[idx] = cn;
        hout[idx] = __float2bfloat16(ov * tanhf(cn));
    }
}

// ---------------------------------------------------------------------------
// Final FC: out[64][256] = h_last(64x1024 bf16) @ fc_w^T + fc_b   (f32 out)
// ---------------------------------------------------------------------------
__global__ __launch_bounds__(128)
void fc_wmma(const __hip_bfloat16* __restrict__ hlast,
             const __hip_bfloat16* __restrict__ wfc,   // [O][H] bf16
             const float* __restrict__ bias,
             float* __restrict__ out) {
    const int lane = threadIdx.x & 31;
    const int wave = threadIdx.x >> 5;
    const int m0 = wave * 16;
    const int j0 = blockIdx.x * 16;
    const int nn = lane & 15;
    const int hi = lane >> 4;
    const int mA = m0 + nn;

    v8f acc = {};
    for (int k0 = 0; k0 < H_DIM; k0 += 32) {
        const int ka = k0 + hi * 8;
        const int kb = k0 + 16 + hi * 8;
        union { uint4 u[2]; v16bf v; } A, Bv;
        const __hip_bfloat16* ar = hlast + (size_t)mA * H_DIM;
        A.u[0] = *(const uint4*)(ar + ka);
        A.u[1] = *(const uint4*)(ar + kb);
        const __hip_bfloat16* wr = wfc + (size_t)(j0 + nn) * H_DIM;
        Bv.u[0] = *(const uint4*)(wr + ka);
        Bv.u[1] = *(const uint4*)(wr + kb);
        acc = __builtin_amdgcn_wmma_f32_16x16x32_bf16(
            false, A.v, false, Bv.v, (short)0, acc, false, false);
    }
    const int j = j0 + nn;
    const float b = bias[j];
#pragma unroll
    for (int r = 0; r < 8; ++r) {
        const int m = m0 + r + hi * 8;
        out[(size_t)m * O_DIM + j] = acc[r] + b;
    }
}

// ---------------------------------------------------------------------------
// Host launcher
// ---------------------------------------------------------------------------
extern "C" void kernel_launch(void* const* d_in, const int* in_sizes, int n_in,
                              void* d_out, int out_size, void* d_ws, size_t ws_size,
                              hipStream_t stream) {
    (void)in_sizes; (void)n_in; (void)out_size; (void)ws_size;

    const float* x     = (const float*)d_in[0];
    const float* W_ih0 = (const float*)d_in[1];
    const float* W_hh0 = (const float*)d_in[2];
    const float* b_ih0 = (const float*)d_in[3];
    const float* b_hh0 = (const float*)d_in[4];
    const float* W_ih1 = (const float*)d_in[5];
    const float* W_hh1 = (const float*)d_in[6];
    const float* b_ih1 = (const float*)d_in[7];
    const float* b_hh1 = (const float*)d_in[8];
    const float* fc_w  = (const float*)d_in[9];
    const float* fc_b  = (const float*)d_in[10];
    float* out = (float*)d_out;

    const int KC0 = I_DIM + H_DIM;   // 1280
    const int KC1 = H_DIM + H_DIM;   // 2048

    uint8_t* ws = (uint8_t*)d_ws;
    size_t off = 0;
    auto wsalloc = [&](size_t bytes) -> uint8_t* {
        uint8_t* p = ws + off;
        off += (bytes + 255) & ~(size_t)255;
        return p;
    };
    __hip_bfloat16* xb    = (__hip_bfloat16*)wsalloc((size_t)B_DIM * T_DIM * I_DIM * 2); // 16 MB
    __hip_bfloat16* W0    = (__hip_bfloat16*)wsalloc((size_t)G4H * KC0 * 2);             // 10 MB (L2-resident)
    __hip_bfloat16* W1    = (__hip_bfloat16*)wsalloc((size_t)G4H * KC1 * 2);             // 16 MB (L2-resident)
    __hip_bfloat16* Wfc   = (__hip_bfloat16*)wsalloc((size_t)O_DIM * H_DIM * 2);
    __hip_bfloat16* hseq0 = (__hip_bfloat16*)wsalloc((size_t)T_DIM * B_DIM * H_DIM * 2); // 64 MB
    __hip_bfloat16* h1a   = (__hip_bfloat16*)wsalloc((size_t)B_DIM * H_DIM * 2);
    __hip_bfloat16* h1b   = (__hip_bfloat16*)wsalloc((size_t)B_DIM * H_DIM * 2);
    __hip_bfloat16* hzero = (__hip_bfloat16*)wsalloc((size_t)B_DIM * H_DIM * 2);
    float*          c0    = (float*)wsalloc((size_t)B_DIM * H_DIM * 4);
    float*          c1    = (float*)wsalloc((size_t)B_DIM * H_DIM * 4);

    // Zero h_{-1} and cell states every call (graph-capture-legal memset nodes).
    (void)hipMemsetAsync(hzero, 0, (size_t)B_DIM * H_DIM * 2, stream);
    (void)hipMemsetAsync(c0,    0, (size_t)B_DIM * H_DIM * 4, stream);
    (void)hipMemsetAsync(c1,    0, (size_t)B_DIM * H_DIM * 4, stream);

    // One-shot conversions / weight packing (bf16, concat-K layout).
    {
        int n = B_DIM * T_DIM * I_DIM;
        cvt_f32_bf16<<<(n + 255) / 256, 256, 0, stream>>>(xb, x, n);
    }
    {
        int n = G4H * KC0;
        pack_wcat_bf16<<<(n + 255) / 256, 256, 0, stream>>>(W0, W_ih0, W_hh0, I_DIM, KC0, n);
    }
    {
        int n = G4H * KC1;
        pack_wcat_bf16<<<(n + 255) / 256, 256, 0, stream>>>(W1, W_ih1, W_hh1, H_DIM, KC1, n);
    }
    {
        int n = O_DIM * H_DIM;
        cvt_f32_bf16<<<(n + 255) / 256, 256, 0, stream>>>(Wfc, fc_w, n);
    }

    const dim3 sgrid(H_DIM / 16);   // 64 workgroups
    const dim3 sblk(128);           // 4 wave32

    // Layer 0: A = [x_t ; h0_{t-1}], hseq0[t] doubles as layer-1 input.
    for (int t = 0; t < T_DIM; ++t) {
        const __hip_bfloat16* hp =
            t ? (hseq0 + (size_t)(t - 1) * B_DIM * H_DIM) : hzero;
        lstm_step_wmma<<<sgrid, sblk, 0, stream>>>(
            xb + (size_t)t * I_DIM, T_DIM * I_DIM, I_DIM,
            hp, W0, KC0, b_ih0, b_hh0, c0,
            hseq0 + (size_t)t * B_DIM * H_DIM);
    }

    // Layer 1: A = [hseq0[t] ; h1_{t-1}] with double-buffered h1.
    __hip_bfloat16* hbuf[2] = { h1a, h1b };
    for (int t = 0; t < T_DIM; ++t) {
        const __hip_bfloat16* hp = t ? hbuf[(t - 1) & 1] : hzero;
        lstm_step_wmma<<<sgrid, sblk, 0, stream>>>(
            hseq0 + (size_t)t * B_DIM * H_DIM, H_DIM, H_DIM,
            hp, W1, KC1, b_ih1, b_hh1, c1,
            hbuf[t & 1]);
    }

    // Final FC on h2[:, T-1, :]   ((T-1)&1 == 1 -> h1b)
    fc_wmma<<<dim3(O_DIM / 16), sblk, 0, stream>>>(hbuf[(T_DIM - 1) & 1], Wfc, fc_b, out);
}
